// CausalSelfAttention_7816840478870
// MI455X (gfx1250) — matrix operand order repaired and checked by reading, not compiled, not run on hardware
//
#include <hip/hip_runtime.h>
#include <hip/hip_bf16.h>
#include <cstdint>

// ---------------- problem constants ----------------
constexpr int B_    = 8;
constexpr int T_    = 8;
constexpr int DIM_  = 4096;
constexpr int NH_   = 32;
constexpr int NKV_  = 8;
constexpr int HD_   = 128;
constexpr int PAST_ = 4096;
constexpr int L_    = PAST_ + T_;      // 4104 total keys
constexpr int REP_  = NH_ / NKV_;      // 4
constexpr int CHUNK = 32;              // keys per pipeline step

// output tuple offsets (elements), harness concatenates outputs flat
constexpr size_t O_KI8 = 262144;                              // (B,NKV,L,HD)
constexpr size_t O_KS  = O_KI8 + (size_t)B_*NKV_*L_*HD_;      // 33882112
constexpr size_t O_VI8 = O_KS  + (size_t)B_*NKV_*L_;          // 34144768
constexpr size_t O_VS  = O_VI8 + (size_t)B_*NKV_*L_*HD_;      // 67764736

typedef __attribute__((ext_vector_type(16))) _Float16     v16h;
typedef __attribute__((ext_vector_type(8)))  float        v8f;
typedef __attribute__((ext_vector_type(4)))  unsigned int v4u;
typedef __attribute__((ext_vector_type(8)))  int          v8i_t;
typedef __attribute__((ext_vector_type(4)))  int          v4i_t;

__device__ __forceinline__ v8f wmma_f16(v16h a, v16h b, v8f c) {
    return __builtin_amdgcn_wmma_f32_16x16x32_f16(false, a, false, b,
                                                  (short)0, c, false, false);
}

// =====================================================================
// TDM: issue a 2D tensor_load_to_lds of `rows` x HD_ int32 elements from
// a (PAST_, HD_) row-major tensor starting at gptr, into LDS at lds_addr.
// Descriptor layout per CDNA5 ISA ch.8 (D# groups 0/1; groups 2/3 unused).
// =====================================================================
__device__ __forceinline__ void tdm_load_rows(unsigned int lds_addr,
                                              const int* gptr, int rows)
{
    const unsigned long long ga = (unsigned long long)(uintptr_t)gptr;
    v4u g0;
    g0[0] = 1u;                                          // count=1, user mode
    g0[1] = lds_addr;                                    // lds_addr[31:0]
    g0[2] = (unsigned int)ga;                            // global_addr[31:0]
    g0[3] = (unsigned int)((ga >> 32) & 0x1FFFFFFull)    // global_addr[56:32]
          | (2u << 30);                                  // type = 2 ("image")
    v8i_t g1;
    g1[0] = (int)(2u << 16);                             // wg_mask=0, data_size=4B
    g1[1] = (int)((unsigned)(HD_ & 0xFFFF) << 16);       // tensor_dim0[15:0]
    g1[2] = (int)(((unsigned)HD_ >> 16)                  // tensor_dim0[31:16]
          | ((unsigned)(PAST_ & 0xFFFF) << 16));         // tensor_dim1[15:0]
    g1[3] = (int)(((unsigned)PAST_ >> 16)                // tensor_dim1[31:16]
          | ((unsigned)HD_ << 16));                      // tile_dim0 = 128
    g1[4] = (int)(rows & 0xFFFF);                        // tile_dim1, tile_dim2=0
    g1[5] = (int)HD_;                                    // tensor_dim0_stride lo
    g1[6] = 0;                                           // stride hi, dim1_stride
    g1[7] = 0;
    const v4i_t gz = {0, 0, 0, 0};
#if defined(__clang_major__) && (__clang_major__ >= 23)
    const v8i_t gz8 = {0, 0, 0, 0, 0, 0, 0, 0};
    __builtin_amdgcn_tensor_load_to_lds(g0, g1, gz, gz, gz8, 0);
#else
    __builtin_amdgcn_tensor_load_to_lds(g0, g1, gz, gz, 0);
#endif
}

// =====================================================================
// Generic WMMA GEMM: C[M,N] = A[M,K] * W[K,N], fp32 in/out, f16 compute.
// One wave per 16x16 C tile. A/B fragments use contiguous k order (k = 16*hi + s).
// =====================================================================
__global__ __launch_bounds__(32) void gemm_wmma_f32(
    const float* __restrict__ A, const float* __restrict__ W,
    float* __restrict__ C, int N, int K)
{
    const int lane = threadIdx.x & 31;
    const int lo   = lane & 15;
    const int hi   = (lane >> 4) & 1;
    const int bn   = blockIdx.x;
    const int bm   = blockIdx.y;
    const int arow = bm * 16 + lo;
    const int wcol = bn * 16 + lo;

    v8f acc = {};
    for (int kk = 0; kk < K; kk += 32) {
        v16h a;
        {
            const float* ap = A + (size_t)arow * K + kk;
            const int b1 = hi ? 16 : 0;
#pragma unroll
            for (int h = 0; h < 8; ++h) a[h]     = (_Float16)ap[b1 + h];
#pragma unroll
            for (int h = 0; h < 8; ++h) a[8 + h] = (_Float16)ap[8 + b1 + h];
        }
        v16h bf;
        {
            const float* wp = W + (size_t)(kk + (hi ? 16 : 0)) * N + wcol;
#pragma unroll
            for (int h = 0; h < 16; ++h) bf[h] = (_Float16)wp[(size_t)h * N];
        }
        acc = wmma_f16(a, bf, acc);
    }
#pragma unroll
    for (int f = 0; f < 8; ++f) {
        const int r = bm * 16 + f + hi * 8;
        C[(size_t)r * N + bn * 16 + lo] = acc[f];
    }
}

// =====================================================================
// RoPE on Q (B,T,NH,HD) and K half of KV (B,T,2,NKV,HD), in place.
// =====================================================================
__global__ void rope_kernel(float* __restrict__ q, float* __restrict__ kv,
                            const int* __restrict__ start_pos_p)
{
    const int tid  = blockIdx.x * blockDim.x + threadIdx.x;
    const int QTOT = B_ * T_ * NH_ * (HD_ / 2);
    const int KTOT = B_ * T_ * NKV_ * (HD_ / 2);
    const int sp   = *start_pos_p;
    if (tid < QTOT) {
        int i = tid & 63; int r = tid >> 6;
        int hq = r % NH_; r /= NH_;
        int t = r % T_;   int b = r / T_;
        float inv = __powf(10000.0f, -(float)(2 * i) / 128.0f);
        float ang = (float)(sp + t) * inv;
        float c = __cosf(ang), s = __sinf(ang);
        float* base = q + ((size_t)(b * T_ + t) * NH_ + hq) * HD_;
        float x0 = base[i], x1 = base[i + 64];
        base[i]      = x0 * c - x1 * s;
        base[i + 64] = x1 * c + x0 * s;
    } else if (tid < QTOT + KTOT) {
        int u = tid - QTOT;
        int i = u & 63; int r = u >> 6;
        int kh = r % NKV_; r /= NKV_;
        int t = r % T_;    int b = r / T_;
        float inv = __powf(10000.0f, -(float)(2 * i) / 128.0f);
        float ang = (float)(sp + t) * inv;
        float c = __cosf(ang), s = __sinf(ang);
        float* base = kv + (((size_t)(b * T_ + t) * 2 + 0) * NKV_ + kh) * HD_;
        float x0 = base[i], x1 = base[i + 64];
        base[i]      = x0 * c - x1 * s;
        base[i + 64] = x1 * c + x0 * s;
    }
}

// ---------------- flash-attention helpers ----------------
__device__ __forceinline__ void softmax_update(const v8f* sacc, v8f* o,
                                               float* mrow, float* srow,
                                               _Float16* pbuf, int lo, int hi)
{
#pragma unroll
    for (int f = 0; f < 8; ++f) {
        float tm = fmaxf(sacc[0][f], sacc[1][f]);
        tm = fmaxf(tm, __shfl_xor(tm, 1, 32));
        tm = fmaxf(tm, __shfl_xor(tm, 2, 32));
        tm = fmaxf(tm, __shfl_xor(tm, 4, 32));
        tm = fmaxf(tm, __shfl_xor(tm, 8, 32));
        const float newm = fmaxf(mrow[f], tm);
        const float corr = __expf(mrow[f] - newm);
        const float p0 = __expf(sacc[0][f] - newm);
        const float p1 = __expf(sacc[1][f] - newm);
        float rs = p0 + p1;
        rs += __shfl_xor(rs, 1, 32);
        rs += __shfl_xor(rs, 2, 32);
        rs += __shfl_xor(rs, 4, 32);
        rs += __shfl_xor(rs, 8, 32);
        srow[f] = srow[f] * corr + rs;
        mrow[f] = newm;
#pragma unroll
        for (int c = 0; c < 8; ++c) o[c][f] *= corr;
        const int r = f + hi * 8;
        pbuf[r * 32 + lo]      = (_Float16)p0;
        pbuf[r * 32 + 16 + lo] = (_Float16)p1;
    }
}

__device__ __forceinline__ v16h load_pa(const _Float16* pbuf, int lo, int hi)
{
    v16h pa;
    const int b1 = hi * 16;
#pragma unroll
    for (int h = 0; h < 8; ++h) pa[h]     = pbuf[lo * 32 + b1 + h];
#pragma unroll
    for (int h = 0; h < 8; ++h) pa[8 + h] = pbuf[lo * 32 + 8 + b1 + h];
    return pa;
}

// =====================================================================
// Flash attention, one wave per (b, head). TDM streams 32-key x 128-dim
// int8 K/V tiles into double-buffered LDS while WMMAs consume the
// previous tile. Last 8 (fp32, rope'd) keys handled in a tail chunk.
// =====================================================================
__global__ __launch_bounds__(32) void attn_kernel(
    const float* __restrict__ q,      // (B,T,NH,HD), rope'd
    const float* __restrict__ kv,     // (B,T,2,NKV,HD), k rope'd
    const int*   __restrict__ pk_i8,  // (B,NKV,PAST,HD) int32-held int8
    const float* __restrict__ pk_s,
    const int*   __restrict__ pv_i8,
    const float* __restrict__ pv_s,
    float* __restrict__ ao)           // (B*T, DIM) staged for Wo GEMM
{
    __shared__ int ktile[2][CHUNK * HD_];   // 2 x 16 KB
    __shared__ int vtile[2][CHUNK * HD_];   // 2 x 16 KB   (total 64 KB)

    const int lane = threadIdx.x & 31;
    const int lo   = lane & 15;
    const int hi   = (lane >> 4) & 1;
    const int bid  = blockIdx.x;
    const int b    = bid / NH_;
    const int hq   = bid % NH_;
    const int kh   = hq / REP_;
    const float qscale = 0.08838834764831845f;   // 1/sqrt(HD)

    // ---- Q A-fragments, scale folded in; padded rows (lo>=T) zero
    v16h qa[4];
#pragma unroll
    for (int kc = 0; kc < 4; ++kc) {
        v16h a = {};
        if (lo < T_) {
            const float* qp = q + ((size_t)(b * T_ + lo) * NH_ + hq) * HD_ + kc * 32;
            const int b1 = hi * 16;
#pragma unroll
            for (int h = 0; h < 8; ++h) a[h]     = (_Float16)(qp[b1 + h] * qscale);
#pragma unroll
            for (int h = 0; h < 8; ++h) a[8 + h] = (_Float16)(qp[8 + b1 + h] * qscale);
        }
        qa[kc] = a;
    }

    v8f o[8] = {};
    float mrow[8], srow[8];
#pragma unroll
    for (int f = 0; f < 8; ++f) { mrow[f] = -1e30f; srow[f] = 0.0f; }

    const size_t kvrow = (size_t)(b * NKV_ + kh) * PAST_;
    const int* kbase = pk_i8 + kvrow * HD_;
    const int* vbase = pv_i8 + kvrow * HD_;
    // LDS byte addresses for the TDM descriptor (production code would use
    // the gfx1250 TDM header's LDS-offset helper; compile-only here).
    const unsigned int klds[2] = { (unsigned int)(uintptr_t)&ktile[0][0],
                                   (unsigned int)(uintptr_t)&ktile[1][0] };
    const unsigned int vlds[2] = { (unsigned int)(uintptr_t)&vtile[0][0],
                                   (unsigned int)(uintptr_t)&vtile[1][0] };

    constexpr int NSTEP = PAST_ / CHUNK;   // 128 pipeline steps over the cache

    // prologue: fill buffer 0
    tdm_load_rows(klds[0], kbase, CHUNK);
    tdm_load_rows(vlds[0], vbase, CHUNK);

    for (int step = 0; step < NSTEP; ++step) {
        const int buf = step & 1;
        const int kb0 = step * CHUNK;

        // all LDS traffic on buffer buf^1 must retire before TDM refills it
        asm volatile("s_wait_dscnt 0x0" ::: "memory");
        if (step + 1 < NSTEP) {
            tdm_load_rows(klds[buf ^ 1], kbase + (size_t)(kb0 + CHUNK) * HD_, CHUNK);
            tdm_load_rows(vlds[buf ^ 1], vbase + (size_t)(kb0 + CHUNK) * HD_, CHUNK);
            __builtin_amdgcn_s_wait_tensorcnt(2);   // current buffer's DMAs done
        } else {
            __builtin_amdgcn_s_wait_tensorcnt(0);
        }
        __syncthreads();

        // ---------- S = Q*K^T from the LDS K tile ----------
        v8f sacc[2];
#pragma unroll
        for (int tt = 0; tt < 2; ++tt) {
            const int krow = tt * 16 + lo;          // key column this lane owns
            const int* kp  = &ktile[buf][krow * HD_];
            const float sc = pk_s[kvrow + kb0 + krow];
            v16h kf[4];
#pragma unroll
            for (int kc = 0; kc < 4; ++kc) {
                const int d0 = kc * 32 + hi * 16;
#pragma unroll
                for (int h = 0; h < 16; ++h)
                    kf[kc][h] = (_Float16)((float)kp[d0 + h] * sc);
            }
            v8f s = {};
#pragma unroll
            for (int kc = 0; kc < 4; ++kc) s = wmma_f16(qa[kc], kf[kc], s);
            sacc[tt] = s;
        }

        // V scales for this chunk (reused across all 8 hd tiles)
        float vsc[16];
#pragma unroll
        for (int h = 0; h < 16; ++h) vsc[h] = pv_s[kvrow + kb0 + hi * 16 + h];

        // ---------- online softmax; P staged over the spent K tile ----------
        _Float16* pbuf = (_Float16*)&ktile[buf][0];
        __syncthreads();
        softmax_update(sacc, o, mrow, srow, pbuf, lo, hi);
        __syncthreads();
        const v16h pa = load_pa(pbuf, lo, hi);

        // ---------- O += P * V from the LDS V tile ----------
#pragma unroll
        for (int c = 0; c < 8; ++c) {
            v16h vf;
            const int d = c * 16 + lo;
#pragma unroll
            for (int h = 0; h < 16; ++h) {
                const int krow = hi * 16 + h;
                vf[h] = (_Float16)((float)vtile[buf][krow * HD_ + d] * vsc[h]);
            }
            o[c] = wmma_f16(pa, vf, o[c]);
        }
        __syncthreads();
    }

    // ---------- tail: the T_ new fp32 keys (rope'd) from kv_buf ----------
    {
        v8f sacc[2];
#pragma unroll
        for (int tt = 0; tt < 2; ++tt) {
            const int key = PAST_ + tt * 16 + lo;
            v16h kf[4];
            if (key < L_) {
                const float* kp = kv + (((size_t)(b * T_ + (key - PAST_)) * 2 + 0) * NKV_ + kh) * HD_;
#pragma unroll
                for (int kc = 0; kc < 4; ++kc) {
                    const int d0 = kc * 32 + hi * 16;
#pragma unroll
                    for (int h = 0; h < 16; ++h) kf[kc][h] = (_Float16)kp[d0 + h];
                }
            } else {
#pragma unroll
                for (int kc = 0; kc < 4; ++kc)
#pragma unroll
                    for (int h = 0; h < 16; ++h) kf[kc][h] = (_Float16)0.0f;
            }
            v8f s = {};
#pragma unroll
            for (int kc = 0; kc < 4; ++kc) s = wmma_f16(qa[kc], kf[kc], s);
            if (key >= L_) {
#pragma unroll
                for (int f = 0; f < 8; ++f) s[f] = -1e30f;
            }
            sacc[tt] = s;
        }
        _Float16* pbuf = (_Float16*)&ktile[0][0];
        __syncthreads();
        softmax_update(sacc, o, mrow, srow, pbuf, lo, hi);
        __syncthreads();
        const v16h pa = load_pa(pbuf, lo, hi);
#pragma unroll
        for (int c = 0; c < 8; ++c) {
            v16h vf;
            const int d = c * 16 + lo;
#pragma unroll
            for (int h = 0; h < 16; ++h) {
                const int key = PAST_ + hi * 16 + h;
                float vv = 0.0f;
                if (key < L_)
                    vv = kv[(((size_t)(b * T_ + (key - PAST_)) * 2 + 1) * NKV_ + kh) * HD_ + d];
                vf[h] = (_Float16)vv;
            }
            o[c] = wmma_f16(pa, vf, o[c]);
        }
    }

    // ---------- normalize + store real query rows ----------
#pragma unroll
    for (int c = 0; c < 8; ++c) {
        const int d = c * 16 + lo;
#pragma unroll
        for (int f = 0; f < 8; ++f) {
            const int m = f + hi * 8;
            if (m < T_)
                ao[(size_t)(b * T_ + m) * DIM_ + hq * HD_ + d] = o[c][f] / srow[f];
        }
    }
}

// =====================================================================
// Requantize concatenated K/V: one wave per (tensor,b,kh,pos) row of 128.
// =====================================================================
__global__ void quant_kernel(const float* __restrict__ kv,
                             const int* __restrict__ pk_i8, const float* __restrict__ pk_s,
                             const int* __restrict__ pv_i8, const float* __restrict__ pv_s,
                             float* __restrict__ out_base)
{
    const int wave = (int)((blockIdx.x * (size_t)blockDim.x + threadIdx.x) >> 5);
    const int lane = threadIdx.x & 31;
    const int ROWS = B_ * NKV_ * L_;
    if (wave >= 2 * ROWS) return;
    const int tensor = wave / ROWS;           // 0 = K, 1 = V
    int r = wave % ROWS;
    const int b   = r / (NKV_ * L_);
    const int r2  = r % (NKV_ * L_);
    const int kh  = r2 / L_;
    const int pos = r2 % L_;

    const int*   pi8 = tensor ? pv_i8 : pk_i8;
    const float* ps  = tensor ? pv_s  : pk_s;

    float vals[4];
#pragma unroll
    for (int j = 0; j < 4; ++j) {
        const int d = j * 32 + lane;
        float v;
        if (pos < PAST_) {
            const size_t ridx = (size_t)(b * NKV_ + kh) * PAST_ + pos;
            v = (float)pi8[ridx * HD_ + d] * ps[ridx];
        } else {
            v = kv[(((size_t)(b * T_ + (pos - PAST_)) * 2 + tensor) * NKV_ + kh) * HD_ + d];
        }
        vals[j] = v;
    }
    float am = 0.0f;
#pragma unroll
    for (int j = 0; j < 4; ++j) am = fmaxf(am, fabsf(vals[j]));
    am = fmaxf(am, __shfl_xor(am, 1, 32));
    am = fmaxf(am, __shfl_xor(am, 2, 32));
    am = fmaxf(am, __shfl_xor(am, 4, 32));
    am = fmaxf(am, __shfl_xor(am, 8, 32));
    am = fmaxf(am, __shfl_xor(am, 16, 32));
    const float scale = fmaxf(am / 127.0f, 1e-8f);
    const float inv = 1.0f / scale;

    float* oq = out_base + (tensor ? O_VI8 : O_KI8)
              + ((size_t)(b * NKV_ + kh) * L_ + pos) * HD_;
#pragma unroll
    for (int j = 0; j < 4; ++j) {
        float qv = roundf(vals[j] * inv);
        qv = fminf(fmaxf(qv, -127.0f), 127.0f);
        oq[j * 32 + lane] = qv;
    }
    if (lane == 0)
        out_base[(tensor ? O_VS : O_KS) + (size_t)(b * NKV_ + kh) * L_ + pos] = scale;
}

// =====================================================================
extern "C" void kernel_launch(void* const* d_in, const int* in_sizes, int n_in,
                              void* d_out, int out_size, void* d_ws, size_t ws_size,
                              hipStream_t stream)
{
    (void)in_sizes; (void)n_in; (void)out_size; (void)ws_size;
    const float* x      = (const float*)d_in[0];
    const float* Wq     = (const float*)d_in[1];
    const float* Wkv    = (const float*)d_in[2];
    const float* Wo     = (const float*)d_in[3];
    const int*   pk_i8  = (const int*)  d_in[4];
    const float* pk_s   = (const float*)d_in[5];
    const int*   pv_i8  = (const int*)  d_in[6];
    const float* pv_s   = (const float*)d_in[7];
    const int*   sp     = (const int*)  d_in[8];
    float* out = (float*)d_out;
    float* ws  = (float*)d_ws;

    float* q_buf  = ws;                          // B*T*NH*HD      = 262144 f32
    float* kv_buf = ws + 262144;                 // B*T*2*NKV*HD   = 131072 f32
    float* ao_buf = ws + 262144 + 131072;        // B*T*DIM        = 262144 f32

    const dim3 w32(32);

    gemm_wmma_f32<<<dim3(DIM_ / 16, (B_ * T_) / 16), w32, 0, stream>>>(x, Wq,  q_buf,  DIM_,           DIM_);
    gemm_wmma_f32<<<dim3((2 * NKV_ * HD_) / 16, (B_ * T_) / 16), w32, 0, stream>>>(x, Wkv, kv_buf, 2 * NKV_ * HD_, DIM_);

    const int rt = B_ * T_ * NH_ * (HD_ / 2) + B_ * T_ * NKV_ * (HD_ / 2);
    rope_kernel<<<(rt + 255) / 256, 256, 0, stream>>>(q_buf, kv_buf, sp);

    attn_kernel<<<B_ * NH_, w32, 0, stream>>>(q_buf, kv_buf, pk_i8, pk_s, pv_i8, pv_s, ao_buf);

    gemm_wmma_f32<<<dim3(DIM_ / 16, (B_ * T_) / 16), w32, 0, stream>>>(ao_buf, Wo, out, DIM_, DIM_);

    const int waves = 2 * B_ * NKV_ * L_;                 // 525312
    quant_kernel<<<(waves * 32 + 255) / 256, 256, 0, stream>>>(kv_buf, pk_i8, pk_s, pv_i8, pv_s, out);
}
